// MagicGAT_14319420965411
// MI455X (gfx1250) — compile-verified, matrix operand order checked by reading, not run on hardware
//
#include <hip/hip_runtime.h>
#include <math.h>

#define NNODES 50000
#define NEDGES 1600000
#define NE2    (NEDGES + NNODES)
#define KDIM   256
#define NEGS   0.2f

typedef __attribute__((ext_vector_type(16))) __bf16 v16bf;
typedef __attribute__((ext_vector_type(8)))  float  v8f;

struct U128 { unsigned x, y, z, w; };           // POD 16-byte chunk
union Frag { U128 u[2]; v16bf v; };             // one WMMA bf16 A/B fragment (8 VGPRs)

static __device__ __forceinline__ unsigned short bf16u(float f) {
  __bf16 h = (__bf16)f;                         // RNE convert
  return __builtin_bit_cast(unsigned short, h);
}
static __device__ __forceinline__ unsigned pack2(float a, float b) {
  return (unsigned)bf16u(a) | ((unsigned)bf16u(b) << 16);
}

// Async global->LDS 16B copy (gfx1250, tracked by ASYNCcnt).
// LDS operand is the 32-bit offset = low bits of the generic shared pointer.
static __device__ __forceinline__ void async_b128(const void* g, const void* l) {
  unsigned lds = (unsigned)(uintptr_t)l;
  unsigned long long ga = (unsigned long long)(uintptr_t)g;
  asm volatile("global_load_async_to_lds_b128 %0, %1, off"
               :: "v"(lds), "v"(ga) : "memory");
}

// ---------------------------------------------------------------------------
// bf16 conversion kernels (run once per matrix, streaming, HBM/L2 bound)
// ---------------------------------------------------------------------------
__global__ void conv_bf16(const float* __restrict__ src,
                          unsigned short* __restrict__ dst, int n4) {
  int i = blockIdx.x * blockDim.x + threadIdx.x;
  if (i >= n4) return;
  float4 v = *(const float4*)(src + (size_t)i * 4);
  *(uint2*)(dst + (size_t)i * 4) = make_uint2(pack2(v.x, v.y), pack2(v.z, v.w));
}

// W[K=256 x Nout] f32  ->  Wt[Nout x 256] bf16 (transposed, row = output col)
__global__ void conv_w_bf16(const float* __restrict__ W,
                            unsigned short* __restrict__ Wt, int Nout) {
  int i = blockIdx.x * blockDim.x + threadIdx.x;
  if (i >= KDIM * Nout) return;
  int k = i / Nout, c = i - k * Nout;
  Wt[(size_t)c * KDIM + k] = bf16u(W[i]);
}

// ---------------------------------------------------------------------------
// Tiled GEMM: C[M x Nout] = A[M x 256] * W[256 x Nout]  (A,Wt already bf16;
// Wt is column-major so both tiles async-copy contiguously into LDS).
// Block = 256 threads = 8 waves (4M x 2N); block tile 128x64; wave tile 32x32
// = 2x2 v_wmma_f32_16x16x32_bf16. Double-buffered async staging (ASYNCcnt).
// ---------------------------------------------------------------------------
__global__ __launch_bounds__(256)
void gat_gemm_wmma(const unsigned short* __restrict__ A,
                   const unsigned short* __restrict__ Wt,
                   float* __restrict__ C, int M, int Nout) {
  // row stride 40 bf16 (80B): keeps 16B alignment for b128 LDS ops, depads banks
  __shared__ unsigned short As[2][128][40];  // A tile 128x32 bf16
  __shared__ unsigned short Bs[2][64][40];   // B tile: Bs[col][k]

  const int bm = blockIdx.x * 128;
  const int bn = blockIdx.y * 64;
  const int tid  = threadIdx.x;
  const int wave = tid >> 5, lane = tid & 31;
  const int wm = wave >> 1, wn = wave & 1;
  const int hlf = lane >> 4, l16 = lane & 15;

  v8f acc[2][2] = {};

  // stage one K-tile: 3 async b128 per thread (2 for A, 1 for B)
  auto stage = [&](int buf, int k0) {
#pragma unroll
    for (int i = 0; i < 2; ++i) {
      int ch = tid + i * 256;                // 0..511 chunks of A tile
      int r = ch >> 2, c = ch & 3;           // row 0..127, 16B chunk 0..3
      int grow = bm + r; if (grow >= M) grow = M - 1;  // clamp: rows >= M never stored
      async_b128(A + (size_t)grow * KDIM + k0 + c * 8, &As[buf][r][c * 8]);
    }
    {
      int r = tid >> 2, c = tid & 3;         // 256 chunks of B tile
      async_b128(Wt + (size_t)(bn + r) * KDIM + k0 + c * 8, &Bs[buf][r][c * 8]);
    }
  };

  stage(0, 0);
  for (int it = 0; it < 8; ++it) {
    const int cur = it & 1;
    if (it < 7) {
      stage(1 - cur, (it + 1) * 32);
      asm volatile("s_wait_asynccnt 0x3" ::: "memory");  // cur tile landed, next in flight
    } else {
      asm volatile("s_wait_asynccnt 0x0" ::: "memory");
    }
    __syncthreads();

    // fragment loads (ds_load_b128) per ISA 7.12.2 bf16 16x32 layout:
    // lanes 0-15: K = 0..7 & 16..23 ; lanes 16-31: K = 8..15 & 24..31
    Frag fa[2], fb[2];
#pragma unroll
    for (int mi = 0; mi < 2; ++mi) {
      int r = wm * 32 + mi * 16 + l16;
      fa[mi].u[0] = *(const U128*)&As[cur][r][hlf * 8];
      fa[mi].u[1] = *(const U128*)&As[cur][r][16 + hlf * 8];
    }
#pragma unroll
    for (int ni = 0; ni < 2; ++ni) {
      int c = wn * 32 + ni * 16 + l16;
      fb[ni].u[0] = *(const U128*)&Bs[cur][c][hlf * 8];
      fb[ni].u[1] = *(const U128*)&Bs[cur][c][16 + hlf * 8];
    }
#pragma unroll
    for (int mi = 0; mi < 2; ++mi)
#pragma unroll
      for (int ni = 0; ni < 2; ++ni)
        acc[mi][ni] = __builtin_amdgcn_wmma_f32_16x16x32_bf16(
            false, fa[mi].v, false, fb[ni].v, (short)0, acc[mi][ni], false, false);

    __syncthreads();   // all waves done reading cur before it is re-staged
  }

  // write out: C/D layout VGPR i -> M = hlf*8 + i, N = l16
#pragma unroll
  for (int mi = 0; mi < 2; ++mi)
#pragma unroll
    for (int ni = 0; ni < 2; ++ni) {
      int c = bn + wn * 32 + ni * 16 + l16;
#pragma unroll
      for (int i = 0; i < 8; ++i) {
        int r = bm + wm * 32 + mi * 16 + hlf * 8 + i;
        if (r < M) C[(size_t)r * Nout + c] = acc[mi][ni][i];
      }
    }
}

// ---------------------------------------------------------------------------
// Per-node attention logits: al[n,h] = dot(hp[n,h,:], a[h,:]) (wave per (n,h))
// ---------------------------------------------------------------------------
__global__ __launch_bounds__(256)
void gat_logits(const float* __restrict__ hp, const float* __restrict__ a_s,
                const float* __restrict__ a_d, float* __restrict__ als,
                float* __restrict__ ald, int nn, int Hh, int Cc) {
  int wid  = (int)((blockIdx.x * blockDim.x + threadIdx.x) >> 5);
  int lane = threadIdx.x & 31;
  if (wid >= nn * Hh) return;
  int n = wid / Hh, h = wid - n * Hh;
  const float* hv  = hp + (size_t)n * Hh * Cc + (size_t)h * Cc;
  const float* asv = a_s + (size_t)h * Cc;
  const float* adv = a_d + (size_t)h * Cc;
  float ss = 0.f, sd = 0.f;
  for (int c = lane; c < Cc; c += 32) {
    float x = hv[c];
    ss += x * asv[c];
    sd += x * adv[c];
  }
#pragma unroll
  for (int off = 16; off > 0; off >>= 1) {
    ss += __shfl_down(ss, off, 32);
    sd += __shfl_down(sd, off, 32);
  }
  if (lane == 0) { als[wid] = ss; ald[wid] = sd; }
}

// ---------------------------------------------------------------------------
// Per-layer init: zero aggregation target, m = -inf, s = 0
// ---------------------------------------------------------------------------
__global__ void gat_init(float* __restrict__ agg, int aggN,
                         float* __restrict__ mx, float* __restrict__ sm, int mN) {
  int i = blockIdx.x * blockDim.x + threadIdx.x;
  if (i < aggN) agg[i] = 0.f;
  if (i < mN) { mx[i] = -INFINITY; sm[i] = 0.f; }
}

static __device__ __forceinline__ void atomicMaxF(float* addr, float val) {
  int* ai = (int*)addr;
  int old = __float_as_int(*addr);
  while (__int_as_float(old) < val) {
    int assumed = old;
    old = atomicCAS(ai, assumed, __float_as_int(val));
    if (old == assumed) break;
  }
}

static __device__ __forceinline__ void edge_pair(const long long* ei, int e,
                                                 int* src, int* dst) {
  if (e < NEDGES) { *src = (int)ei[e]; *dst = (int)ei[NEDGES + e]; }
  else            { *src = *dst = e - NEDGES; }   // self loop
}

// Segment max of leaky-relu'd edge logits (thread per edge)
__global__ void gat_edge_max(const long long* __restrict__ ei,
                             const float* __restrict__ als,
                             const float* __restrict__ ald,
                             float* __restrict__ mx, int Hh) {
  int e = blockIdx.x * blockDim.x + threadIdx.x;
  if (e >= NE2) return;
  int src, dst; edge_pair(ei, e, &src, &dst);
  for (int h = 0; h < Hh; ++h) {
    float v = als[src * Hh + h] + ald[dst * Hh + h];
    v = v > 0.f ? v : v * NEGS;
    atomicMaxF(&mx[dst * Hh + h], v);
  }
}

// Segment sum of exp(logit - max) (thread per edge)
__global__ void gat_edge_sum(const long long* __restrict__ ei,
                             const float* __restrict__ als,
                             const float* __restrict__ ald,
                             const float* __restrict__ mx,
                             float* __restrict__ sm, int Hh) {
  int e = blockIdx.x * blockDim.x + threadIdx.x;
  if (e >= NE2) return;
  int src, dst; edge_pair(ei, e, &src, &dst);
  for (int h = 0; h < Hh; ++h) {
    float v = als[src * Hh + h] + ald[dst * Hh + h];
    v = v > 0.f ? v : v * NEGS;
    atomicAdd(&sm[dst * Hh + h], __expf(v - mx[dst * Hh + h]));
  }
}

// Weighted scatter-add aggregation (wave per edge; lanes stride features).
// Gathers + GLOBAL_ATOMIC_ADD_F32 land in the 192MB L2 (node state = 51MB).
__global__ __launch_bounds__(256)
void gat_aggregate(const long long* __restrict__ ei,
                   const float* __restrict__ als, const float* __restrict__ ald,
                   const float* __restrict__ mx, const float* __restrict__ sm,
                   const float* __restrict__ hp, float* __restrict__ agg,
                   int Hh, int Cc) {
  int wid  = (int)((blockIdx.x * blockDim.x + threadIdx.x) >> 5);
  int lane = threadIdx.x & 31;
  if (wid >= NE2) return;
  int src, dst; edge_pair(ei, wid, &src, &dst);
  float coef[4];
  for (int h = 0; h < Hh; ++h) {
    float v = als[src * Hh + h] + ald[dst * Hh + h];
    v = v > 0.f ? v : v * NEGS;
    coef[h] = __expf(v - mx[dst * Hh + h]) / sm[dst * Hh + h];
  }
  const int HC = Hh * Cc;
  const float* hsrc = hp + (size_t)src * HC;
  float* adst = agg + (size_t)dst * HC;
  for (int idx = lane; idx < HC; idx += 32) {
    int h = idx / Cc;
    atomicAdd(&adst[idx], coef[h] * hsrc[idx]);
  }
}

// Epilogue: out[n,c] += bias[c] (+ res[n,c]) (+ bias2[c])
__global__ void gat_post(float* __restrict__ out, const float* __restrict__ bias,
                         const float* __restrict__ res, const float* __restrict__ bias2,
                         int nn, int HC) {
  int i = blockIdx.x * blockDim.x + threadIdx.x;
  if (i >= nn * HC) return;
  int c = i % HC;
  float v = out[i] + bias[c];
  if (res)   v += res[i];
  if (bias2) v += bias2[c];
  out[i] = v;
}

// ---------------------------------------------------------------------------
extern "C" void kernel_launch(void* const* d_in, const int* in_sizes, int n_in,
                              void* d_out, int out_size, void* d_ws, size_t ws_size,
                              hipStream_t stream) {
  const float*     x   = (const float*)d_in[0];
  const long long* ei  = (const long long*)d_in[1];
  const float* W0  = (const float*)d_in[2];
  const float* as0 = (const float*)d_in[3];
  const float* ad0 = (const float*)d_in[4];
  const float* b0  = (const float*)d_in[5];
  const float* W1  = (const float*)d_in[6];
  const float* as1 = (const float*)d_in[7];
  const float* ad1 = (const float*)d_in[8];
  const float* b1  = (const float*)d_in[9];
  const float* W2  = (const float*)d_in[10];
  const float* as2 = (const float*)d_in[11];
  const float* ad2 = (const float*)d_in[12];
  const float* b2  = (const float*)d_in[13];
  const float* Wl  = (const float*)d_in[14];
  const float* bl  = (const float*)d_in[15];
  float* out = (float*)d_out;

  // workspace partition
  float* hp  = (float*)d_ws;                       // [N,256] current projection (f32)
  float* h0  = hp  + (size_t)NNODES * 256;         // [N,256]
  float* h1  = h0  + (size_t)NNODES * 256;         // [N,256]
  float* lin = h1  + (size_t)NNODES * 256;         // [N,128]
  float* als = lin + (size_t)NNODES * 128;         // [N,4]
  float* ald = als + (size_t)NNODES * 4;           // [N,4]
  float* mx  = ald + (size_t)NNODES * 4;           // [N,4]
  float* sm  = mx  + (size_t)NNODES * 4;           // [N,4]
  unsigned short* Abf = (unsigned short*)(sm + (size_t)NNODES * 4);  // [N,256] bf16
  unsigned short* W0t = Abf + (size_t)NNODES * 256;                  // [256,256] bf16^T
  unsigned short* W1t = W0t + 256 * 256;
  unsigned short* W2t = W1t + 256 * 256;                             // [128,256]
  unsigned short* Wlt = W2t + 256 * 256;                             // [128,256]

  const dim3 blk(256);
  const dim3 gemm_g256((NNODES + 127) / 128, 4);   // Nout=256
  const dim3 gemm_g128((NNODES + 127) / 128, 2);   // Nout=128
  const int eg   = (NE2 + 255) / 256;              // thread-per-edge grids
  const int eag  = (NE2 + 7) / 8;                  // wave-per-edge grids
  const int cg   = (NNODES * 256 / 4 + 255) / 256; // activation bf16 convert grid
  const int ng   = (NNODES * 256 + 255) / 256;

  // weight conversions (transposed bf16), done once up front
  conv_w_bf16<<<(256 * 256 + 255) / 256, blk, 0, stream>>>(W0, W0t, 256);
  conv_w_bf16<<<(256 * 256 + 255) / 256, blk, 0, stream>>>(W1, W1t, 256);
  conv_w_bf16<<<(256 * 128 + 255) / 256, blk, 0, stream>>>(W2, W2t, 128);
  conv_w_bf16<<<(256 * 128 + 255) / 256, blk, 0, stream>>>(Wl, Wlt, 128);

  // ---------------- layer 0: x -> h0 (H=4, C=64) ----------------
  conv_bf16<<<cg, blk, 0, stream>>>(x, Abf, NNODES * 256 / 4);
  gat_gemm_wmma<<<gemm_g256, blk, 0, stream>>>(Abf, W0t, hp, NNODES, 256);
  gat_logits<<<(NNODES * 4 + 7) / 8, blk, 0, stream>>>(hp, as0, ad0, als, ald, NNODES, 4, 64);
  gat_init<<<ng, blk, 0, stream>>>(h0, NNODES * 256, mx, sm, NNODES * 4);
  gat_edge_max<<<eg, blk, 0, stream>>>(ei, als, ald, mx, 4);
  gat_edge_sum<<<eg, blk, 0, stream>>>(ei, als, ald, mx, sm, 4);
  gat_aggregate<<<eag, blk, 0, stream>>>(ei, als, ald, mx, sm, hp, h0, 4, 64);
  gat_post<<<ng, blk, 0, stream>>>(h0, b0, nullptr, nullptr, NNODES, 256);

  // ---------------- layer 1: h0 -> h1 (+h0 residual) ----------------
  conv_bf16<<<cg, blk, 0, stream>>>(h0, Abf, NNODES * 256 / 4);
  gat_gemm_wmma<<<gemm_g256, blk, 0, stream>>>(Abf, W1t, hp, NNODES, 256);
  gat_logits<<<(NNODES * 4 + 7) / 8, blk, 0, stream>>>(hp, as1, ad1, als, ald, NNODES, 4, 64);
  gat_init<<<ng, blk, 0, stream>>>(h1, NNODES * 256, mx, sm, NNODES * 4);
  gat_edge_max<<<eg, blk, 0, stream>>>(ei, als, ald, mx, 4);
  gat_edge_sum<<<eg, blk, 0, stream>>>(ei, als, ald, mx, sm, 4);
  gat_aggregate<<<eag, blk, 0, stream>>>(ei, als, ald, mx, sm, hp, h1, 4, 64);
  gat_post<<<ng, blk, 0, stream>>>(h1, b1, h0, nullptr, NNODES, 256);

  // ---------------- layer 2: h1 -> out (H=1, C=128) + (h1@Wl + bl) ----------------
  conv_bf16<<<cg, blk, 0, stream>>>(h1, Abf, NNODES * 256 / 4);   // one convert, two GEMMs
  gat_gemm_wmma<<<gemm_g128, blk, 0, stream>>>(Abf, Wlt, lin, NNODES, 128);
  gat_gemm_wmma<<<gemm_g128, blk, 0, stream>>>(Abf, W2t, hp, NNODES, 128);
  gat_logits<<<(NNODES * 1 + 7) / 8, blk, 0, stream>>>(hp, as2, ad2, als, ald, NNODES, 1, 128);
  gat_init<<<(NNODES * 128 + 255) / 256, blk, 0, stream>>>(out, NNODES * 128, mx, sm, NNODES * 1);
  gat_edge_max<<<eg, blk, 0, stream>>>(ei, als, ald, mx, 1);
  gat_edge_sum<<<eg, blk, 0, stream>>>(ei, als, ald, mx, sm, 1);
  gat_aggregate<<<eag, blk, 0, stream>>>(ei, als, ald, mx, sm, hp, out, 1, 128);
  gat_post<<<(NNODES * 128 + 255) / 256, blk, 0, stream>>>(out, b2, lin, bl, NNODES, 128);
}
